// LiteSparseMoERouter_87875030876704
// MI455X (gfx1250) — matrix operand order repaired and checked by reading, not compiled, not run on hardware
//
#include <hip/hip_runtime.h>
#include <hip/hip_bf16.h>
#include <math.h>

#define NT   16384   // tokens
#define DIMD 2048    // model dim
#define NE   8       // experts
#define HD   4096    // hidden dim (2*DIMD)
#define TOPK 2

// ---- GEMM tiling: block = 8 waves (2 M x 4 N), tile 32 x 256, K-panel 64 ----
#define KC     64                    // k elements per TDM panel
#define BCOLS  256                   // weight columns per block panel
#define ROWB   72                    // bf16 per LDS row: 64 data + 8 pad (bank-spread)
#define PANEL_ELEMS (BCOLS * ROWB)
#define PANEL_BYTES (PANEL_ELEMS * 2)

typedef __bf16 bf16_t;
typedef __bf16 v16bf  __attribute__((ext_vector_type(16)));
typedef __bf16 bf16x8 __attribute__((ext_vector_type(8)));
typedef float  v8f    __attribute__((ext_vector_type(8)));
typedef unsigned int v4u __attribute__((ext_vector_type(4)));
typedef int v4i __attribute__((ext_vector_type(4)));
typedef int v8i __attribute__((ext_vector_type(8)));

union BFrag { v16bf v; bf16x8 h[2]; };

// ---------------------------------------------------------------- init ------
__global__ void k_zero_small(int* counts, int* cursors, float* usage) {
  int t = threadIdx.x;
  if (t < NE) { counts[t] = 0; cursors[t] = 0; usage[t] = 0.f; }
}

__global__ void k_zero_out(float4* __restrict__ out) {
  size_t i = (size_t)blockIdx.x * blockDim.x + threadIdx.x;  // NT*DIMD/4 exact
  out[i] = make_float4(0.f, 0.f, 0.f, 0.f);
}

// -------------------------------------------------------------- router ------
// One wave32 per token: 8 dot products of length 2048, shuffle-reduced.
__global__ __launch_bounds__(256)
void k_router(const float* __restrict__ x, const float* __restrict__ Wr,
              int* __restrict__ top_e, float* __restrict__ top_g,
              float* __restrict__ usage_sum, int* __restrict__ counts) {
  const int lane = threadIdx.x & 31;
  const int t = blockIdx.x * (blockDim.x >> 5) + (threadIdx.x >> 5);
  if (t >= NT) return;
  float acc[NE];
#pragma unroll
  for (int e = 0; e < NE; ++e) acc[e] = 0.f;
  const float* xr = x + (size_t)t * DIMD;
  for (int k = lane; k < DIMD; k += 32) {
    const float xv = xr[k];
    const float* wr = Wr + (size_t)k * NE;
#pragma unroll
    for (int e = 0; e < NE; ++e) acc[e] = fmaf(xv, wr[e], acc[e]);
  }
#pragma unroll
  for (int e = 0; e < NE; ++e)
#pragma unroll
    for (int off = 16; off > 0; off >>= 1)
      acc[e] += __shfl_xor(acc[e], off, 32);

  if (lane == 0) {
    float mx = acc[0];
#pragma unroll
    for (int e = 1; e < NE; ++e) mx = fmaxf(mx, acc[e]);
    float p[NE]; float s = 0.f;
#pragma unroll
    for (int e = 0; e < NE; ++e) { p[e] = __expf(acc[e] - mx); s += p[e]; }
    const float inv = 1.f / s;
#pragma unroll
    for (int e = 0; e < NE; ++e) p[e] *= inv;
    // top-2 (ties -> lowest index, matching lax.top_k)
    int e0 = 0;
#pragma unroll
    for (int e = 1; e < NE; ++e) if (p[e] > p[e0]) e0 = e;
    int e1 = (e0 == 0) ? 1 : 0;
#pragma unroll
    for (int e = 0; e < NE; ++e) if (e != e0 && p[e] > p[e1]) e1 = e;
    const float gs = p[e0] + p[e1];
    top_e[t * 2 + 0] = e0;  top_g[t * 2 + 0] = p[e0] / gs;
    top_e[t * 2 + 1] = e1;  top_g[t * 2 + 1] = p[e1] / gs;
#pragma unroll
    for (int e = 0; e < NE; ++e)
      __hip_atomic_fetch_add(&usage_sum[e], p[e], __ATOMIC_RELAXED,
                             __HIP_MEMORY_SCOPE_AGENT);
    atomicAdd(&counts[e0], 1);
    atomicAdd(&counts[e1], 1);
  }
}

// -------------------------------------------- offsets + load-balance loss ---
__global__ void k_offsets_loss(const int* __restrict__ counts,
                               const float* __restrict__ usage_sum,
                               int* __restrict__ offsets, int* __restrict__ cursors,
                               float* __restrict__ loss_out) {
  if (threadIdx.x == 0 && blockIdx.x == 0) {
    int off = 0; float loss = 0.f;
    for (int e = 0; e < NE; ++e) {
      offsets[e] = off; off += counts[e]; cursors[e] = 0;
      const float usage = usage_sum[e] * (1.f / NT);
      const float sel   = (float)counts[e] * (1.f / (NT * TOPK));
      loss += usage * sel;
    }
    loss_out[0] = (float)NE * loss;
  }
}

// ------------------------------------------------- compact slot assignment --
__global__ void k_assign(const int* __restrict__ top_e, const float* __restrict__ top_g,
                         const int* __restrict__ offsets, int* __restrict__ cursors,
                         int* __restrict__ tok_list, float* __restrict__ gate_list) {
  const int t = blockIdx.x * blockDim.x + threadIdx.x;
  if (t >= NT) return;
#pragma unroll
  for (int j = 0; j < TOPK; ++j) {
    const int e = top_e[t * 2 + j];
    const int i = atomicAdd(&cursors[e], 1);
    const int pos = offsets[e] + i;
    tok_list[pos]  = t;
    gate_list[pos] = top_g[t * 2 + j];
  }
}

// ---------------------------------- LDS-tiled transpose + fp32->bf16 --------
// src: [R, C] fp32 (per expert), dst: [C, R] bf16. Coalesced in and out.
__global__ __launch_bounds__(256)
void k_transpose_cvt(const float* __restrict__ src, bf16_t* __restrict__ dst,
                     int R, int C) {
  __shared__ float tile[32][33];
  const size_t esz = (size_t)R * C;
  const float* s = src + (size_t)blockIdx.z * esz;
  bf16_t* d = dst + (size_t)blockIdx.z * esz;
  const int c0 = blockIdx.x * 32, r0 = blockIdx.y * 32;
#pragma unroll
  for (int j = threadIdx.y; j < 32; j += 8)
    tile[j][threadIdx.x] = s[(size_t)(r0 + j) * C + c0 + threadIdx.x];
  __syncthreads();
#pragma unroll
  for (int j = threadIdx.y; j < 32; j += 8)
    d[(size_t)(c0 + j) * R + r0 + threadIdx.x] = (bf16_t)tile[threadIdx.x][j];
}

// ------------------------------------------------------------ GEMM helpers --
__device__ __forceinline__ v16bf load_a_frag_f32(const float* p) {
  // 8 fp32 at p (k..k+7) and 8 fp32 at p+16 (k+16..k+23), converted to bf16
  const float4* f0 = (const float4*)p;
  const float4* f1 = (const float4*)(p + 16);
  float4 a = f0[0], b = f0[1], c = f1[0], d = f1[1];
  v16bf r;
  r[0] = (bf16_t)a.x;  r[1] = (bf16_t)a.y;  r[2]  = (bf16_t)a.z;  r[3]  = (bf16_t)a.w;
  r[4] = (bf16_t)b.x;  r[5] = (bf16_t)b.y;  r[6]  = (bf16_t)b.z;  r[7]  = (bf16_t)b.w;
  r[8] = (bf16_t)c.x;  r[9] = (bf16_t)c.y;  r[10] = (bf16_t)c.z;  r[11] = (bf16_t)c.w;
  r[12] = (bf16_t)d.x; r[13] = (bf16_t)d.y; r[14] = (bf16_t)d.z;  r[15] = (bf16_t)d.w;
  return r;
}

__device__ __forceinline__ float gelu_exact(float v) {
  return 0.5f * v * (1.f + erff(v * 0.70710678118654752f));
}

// -------------------- Tensor Data Mover: 2D weight panel -> LDS -------------
// Loads a BCOLS x KC bf16 tile (row stride = row_stride_elems in memory) into
// LDS at lds_off, inserting 16B of pad per 128B row so ds_load rows land on
// distinct banks. D# packing per CDNA5 ISA.
__device__ __forceinline__ void tdm_load_panel(unsigned lds_off, const void* gptr,
                                               unsigned row_stride_elems) {
  const unsigned long long ga = (unsigned long long)(uintptr_t)gptr;
  v4u g0;
  g0[0] = 1u;                                             // count=1 (valid D#)
  g0[1] = lds_off;                                        // lds_addr
  g0[2] = (unsigned)(ga & 0xFFFFFFFFull);                 // global_addr[31:0]
  g0[3] = (unsigned)((ga >> 32) & 0x1FFFFFFull)           // global_addr[56:32]
        | (2u << 30);                                     // type=2 (image)
  const unsigned td0 = 1u << 20, td1 = 1u << 20;          // large dims: no clip
  v8i g1;
  g1[0] = (int)((1u << 16)      // data_size = 2 bytes
              | (1u << 20)      // pad_enable
              | (4u << 22)      // pad_interval: 16 x 8B = 128B rows
              | (3u << 25));    // pad_amount: 4 DWORDs = 16B
  g1[1] = (int)((td0 & 0xFFFFu) << 16);                   // tensor_dim0 lo
  g1[2] = (int)((td0 >> 16) | ((td1 & 0xFFFFu) << 16));   // dim0 hi | dim1 lo
  g1[3] = (int)((td1 >> 16) | ((unsigned)KC << 16));      // dim1 hi | tile_dim0
  g1[4] = (int)BCOLS;                                     // tile_dim1 (| tile_dim2=0)
  g1[5] = (int)row_stride_elems;                          // tensor_dim0_stride lo32
  g1[6] = 0; g1[7] = 0;
  const v4i gz = {0, 0, 0, 0};
#if __clang_major__ >= 23
  const v8i gz8 = {0, 0, 0, 0, 0, 0, 0, 0};
  __builtin_amdgcn_tensor_load_to_lds(g0, g1, gz, gz, gz8, 0);
#else
  __builtin_amdgcn_tensor_load_to_lds(g0, g1, gz, gz, 0);
#endif
}

// --------------------------------------------------------------- GEMM 1 -----
// h[slot, 0:HD] = gelu( x[tok(slot), :] @ W1T[e]^T + b1[e] ), bf16 out.
__global__ __launch_bounds__(256)
void k_gemm1(const float* __restrict__ x, const bf16_t* __restrict__ W1T,
             const float* __restrict__ b1,
             const int* __restrict__ tok_list, const int* __restrict__ counts,
             const int* __restrict__ offsets, bf16_t* __restrict__ hbuf) {
  __shared__ bf16_t sh[2 * PANEL_ELEMS];

  const int e = blockIdx.z;
  const int cnt = counts[e];
  const int m0 = blockIdx.x * 32;
  if (m0 >= cnt) return;                       // uniform per block
  const int base = offsets[e];
  const int lane = threadIdx.x & 31;
  const int wave = threadIdx.x >> 5;
  const int wm = wave >> 2, wn = wave & 3;
  const int row0 = m0 + wm * 16;
  const int lm = lane & 15, lh = lane >> 4;

  int mloc = row0 + lm; if (mloc > cnt - 1) mloc = cnt - 1;
  const int tok = tok_list[base + mloc];
  const float* aptr = x + (size_t)tok * DIMD + lh * 8;
  const bf16_t* wrow = W1T + ((size_t)e * HD + (size_t)blockIdx.y * BCOLS) * DIMD;

  if (wave == 0) tdm_load_panel(0, wrow, DIMD);          // panel 0 -> buf 0

  v8f acc[4] = {};
  const int NP = DIMD / KC;                               // 32 panels
  for (int kp = 0; kp < NP; ++kp) {
    const int buf = kp & 1;
    if (wave == 0) {
      if (kp + 1 < NP) {
        tdm_load_panel((unsigned)((buf ^ 1) * PANEL_BYTES),
                       wrow + (size_t)(kp + 1) * KC, DIMD);
        __builtin_amdgcn_s_wait_tensorcnt(1);   // panel kp complete
      } else {
        __builtin_amdgcn_s_wait_tensorcnt(0);   // last panel complete
      }
    }
    __syncthreads();                                      // publish panel kp
    const bf16_t* pb = sh + buf * PANEL_ELEMS;
#pragma unroll
    for (int sub = 0; sub < 2; ++sub) {
      const v16bf a = load_a_frag_f32(aptr + kp * KC + sub * 32);
#pragma unroll
      for (int nt = 0; nt < 4; ++nt) {
        const int col = wn * 64 + nt * 16 + lm;
        const bf16_t* q = pb + col * ROWB + sub * 32 + lh * 8;
        BFrag f; f.h[0] = ((const bf16x8*)q)[0]; f.h[1] = ((const bf16x8*)(q + 16))[0];
        acc[nt] = __builtin_amdgcn_wmma_f32_16x16x32_bf16(
            false, a, false, f.v, (short)0, acc[nt], false, false);
      }
    }
    __syncthreads();                                      // done reading buf
  }

  const int n0 = blockIdx.y * BCOLS + wn * 64;
#pragma unroll
  for (int v = 0; v < 8; ++v) {
    const int rl = row0 + 8 * lh + v;
    if (rl < cnt) {
      const size_t slot = (size_t)(base + rl);
#pragma unroll
      for (int nt = 0; nt < 4; ++nt) {
        const int n = n0 + nt * 16 + lm;
        float val = acc[nt][v] + b1[e * HD + n];
        hbuf[slot * HD + n] = (bf16_t)gelu_exact(val);
      }
    }
  }
}

// --------------------------------------------------------------- GEMM 2 -----
// out[tok(slot), :] += gate(slot) * ( h[slot, :] @ W2T[e]^T + b2[e] )
__global__ __launch_bounds__(256)
void k_gemm2(const bf16_t* __restrict__ hbuf, const bf16_t* __restrict__ W2T,
             const float* __restrict__ b2,
             const int* __restrict__ tok_list, const float* __restrict__ gate_list,
             const int* __restrict__ counts, const int* __restrict__ offsets,
             float* __restrict__ out) {
  __shared__ bf16_t sh[2 * PANEL_ELEMS];

  const int e = blockIdx.z;
  const int cnt = counts[e];
  const int m0 = blockIdx.x * 32;
  if (m0 >= cnt) return;
  const int base = offsets[e];
  const int lane = threadIdx.x & 31;
  const int wave = threadIdx.x >> 5;
  const int wm = wave >> 2, wn = wave & 3;
  const int row0 = m0 + wm * 16;
  const int lm = lane & 15, lh = lane >> 4;

  int mloc = row0 + lm; if (mloc > cnt - 1) mloc = cnt - 1;
  const bf16_t* aptr = hbuf + (size_t)(base + mloc) * HD + lh * 8;
  const bf16_t* wrow = W2T + ((size_t)e * DIMD + (size_t)blockIdx.y * BCOLS) * HD;

  if (wave == 0) tdm_load_panel(0, wrow, HD);

  v8f acc[4] = {};
  const int NP = HD / KC;                                 // 64 panels
  for (int kp = 0; kp < NP; ++kp) {
    const int buf = kp & 1;
    if (wave == 0) {
      if (kp + 1 < NP) {
        tdm_load_panel((unsigned)((buf ^ 1) * PANEL_BYTES),
                       wrow + (size_t)(kp + 1) * KC, HD);
        __builtin_amdgcn_s_wait_tensorcnt(1);
      } else {
        __builtin_amdgcn_s_wait_tensorcnt(0);
      }
    }
    __syncthreads();
    const bf16_t* pb = sh + buf * PANEL_ELEMS;
#pragma unroll
    for (int sub = 0; sub < 2; ++sub) {
      const int kk = kp * KC + sub * 32;
      BFrag a; a.h[0] = ((const bf16x8*)(aptr + kk))[0];
               a.h[1] = ((const bf16x8*)(aptr + kk + 16))[0];
#pragma unroll
      for (int nt = 0; nt < 4; ++nt) {
        const int col = wn * 64 + nt * 16 + lm;
        const bf16_t* q = pb + col * ROWB + sub * 32 + lh * 8;
        BFrag f; f.h[0] = ((const bf16x8*)q)[0]; f.h[1] = ((const bf16x8*)(q + 16))[0];
        acc[nt] = __builtin_amdgcn_wmma_f32_16x16x32_bf16(
            false, a.v, false, f.v, (short)0, acc[nt], false, false);
      }
    }
    __syncthreads();
  }

  const int n0 = blockIdx.y * BCOLS + wn * 64;
#pragma unroll
  for (int v = 0; v < 8; ++v) {
    const int rl = row0 + 8 * lh + v;
    if (rl < cnt) {
      const int slot = base + rl;
      const int tok = tok_list[slot];
      const float g = gate_list[slot];
      float* orow = out + (size_t)tok * DIMD;
#pragma unroll
      for (int nt = 0; nt < 4; ++nt) {
        const int n = n0 + nt * 16 + lm;
        const float val = (acc[nt][v] + b2[e * DIMD + n]) * g;
        __hip_atomic_fetch_add(&orow[n], val, __ATOMIC_RELAXED,
                               __HIP_MEMORY_SCOPE_AGENT);
      }
    }
  }
}

// ---------------------------------------------------------------- launch ----
extern "C" void kernel_launch(void* const* d_in, const int* in_sizes, int n_in,
                              void* d_out, int out_size, void* d_ws, size_t ws_size,
                              hipStream_t stream) {
  const float* x  = (const float*)d_in[0];  // [NT, DIMD]
  const float* Wr = (const float*)d_in[1];  // [DIMD, NE]
  const float* W1 = (const float*)d_in[2];  // [NE, DIMD, HD]
  const float* b1 = (const float*)d_in[3];  // [NE, HD]
  const float* W2 = (const float*)d_in[4];  // [NE, HD, DIMD]
  const float* b2 = (const float*)d_in[5];  // [NE, DIMD]
  float* out = (float*)d_out;               // [NT*DIMD] output ++ [1] loss

  char* ws = (char*)d_ws;
  int*   counts    = (int*)(ws + 0);
  int*   cursors   = (int*)(ws + 32);
  float* usage     = (float*)(ws + 64);
  int*   offsets   = (int*)(ws + 96);
  size_t o = 256;
  int*   top_e     = (int*)(ws + o);   o += (size_t)NT * 2 * sizeof(int);
  float* top_g     = (float*)(ws + o); o += (size_t)NT * 2 * sizeof(float);
  int*   tok_list  = (int*)(ws + o);   o += (size_t)NT * 2 * sizeof(int);
  float* gate_list = (float*)(ws + o); o += (size_t)NT * 2 * sizeof(float);
  o = (o + 255) & ~(size_t)255;
  bf16_t* W1T  = (bf16_t*)(ws + o);    o += (size_t)NE * DIMD * HD * sizeof(bf16_t);
  bf16_t* W2T  = (bf16_t*)(ws + o);    o += (size_t)NE * DIMD * HD * sizeof(bf16_t);
  bf16_t* hbuf = (bf16_t*)(ws + o);    // [NT*TOPK, HD] bf16

  k_zero_small<<<1, 32, 0, stream>>>(counts, cursors, usage);
  k_zero_out<<<(NT * DIMD / 4) / 256, 256, 0, stream>>>((float4*)out);

  k_router<<<NT / 8, 256, 0, stream>>>(x, Wr, top_e, top_g, usage, counts);
  k_offsets_loss<<<1, 1, 0, stream>>>(counts, usage, offsets, cursors,
                                      out + (size_t)NT * DIMD);
  k_assign<<<(NT + 255) / 256, 256, 0, stream>>>(top_e, top_g, offsets, cursors,
                                                 tok_list, gate_list);

  // W1 [e][DIMD][HD] -> W1T [e][HD][DIMD];  W2 [e][HD][DIMD] -> W2T [e][DIMD][HD]
  k_transpose_cvt<<<dim3(HD / 32, DIMD / 32, NE), dim3(32, 8), 0, stream>>>(W1, W1T, DIMD, HD);
  k_transpose_cvt<<<dim3(DIMD / 32, HD / 32, NE), dim3(32, 8), 0, stream>>>(W2, W2T, HD, DIMD);

  k_gemm1<<<dim3(NT / 32, HD / 256, NE), 256, 0, stream>>>(
      x, W1T, b1, tok_list, counts, offsets, hbuf);
  k_gemm2<<<dim3(NT / 32, DIMD / 256, NE), 256, 0, stream>>>(
      hbuf, W2T, b2, tok_list, gate_list, counts, offsets, out);
}
